// LongformerEncoder_32255204393618
// MI455X (gfx1250) — matrix operand order repaired and checked by reading, not compiled, not run on hardware
//
#include <hip/hip_runtime.h>
#include <hip/hip_bf16.h>
#include <stdint.h>
#include <stddef.h>

// ---------------------------------------------------------------------------
// Longformer encoder layer for MI455X (gfx1250, wave32, WMMA).
// B=1, S=4096, D=768, H=12, DH=64, W=256, DFF=3072, C=S/W=16.
// All GEMMs in bf16 via v_wmma_f32_16x16x32_bf16; fp32 residual/LN spine.
// GEMM A-tiles staged with GLOBAL_LOAD_ASYNC_TO_LDS_B128 + double buffering
// (one barrier per K-tile, copies overlap WMMA); sync fallback guarded.
// ---------------------------------------------------------------------------

typedef __attribute__((ext_vector_type(16))) __bf16 bf16x16;
typedef __attribute__((ext_vector_type(8)))  __bf16 bf16x8;
typedef __attribute__((ext_vector_type(8)))  float  f32x8;
typedef __attribute__((ext_vector_type(4)))  int    i32x4;

#define DEV __device__ __forceinline__

#if __has_builtin(__builtin_amdgcn_global_load_async_to_lds_b128) && \
    __has_builtin(__builtin_amdgcn_s_wait_asynccnt)
#define HAVE_ASYNC_LDS 1
typedef __attribute__((address_space(1))) void  gas_void;
typedef __attribute__((address_space(3))) void  las_void;
typedef __attribute__((address_space(1))) i32x4 gas_i32x4;
typedef __attribute__((address_space(3))) i32x4 las_i32x4;
#else
#define HAVE_ASYNC_LDS 0
#endif

// async copy of 16 bytes global -> LDS (per active lane); ASYNCcnt-tracked
DEV void async_cp16(const void* g, void* l) {
#if HAVE_ASYNC_LDS
  __builtin_amdgcn_global_load_async_to_lds_b128(
      (gas_i32x4*)(gas_void*)(void*)(uintptr_t)g,
      (las_i32x4*)(las_void*)(void*)(uintptr_t)l, 0, 0);
#else
  *(bf16x8*)l = *(const bf16x8*)g;
#endif
}
DEV void async_fence() {
#if HAVE_ASYNC_LDS
  __builtin_amdgcn_s_wait_asynccnt(0);
#endif
}

DEV f32x8 wmma_bf16(bf16x16 a, bf16x16 b, f32x8 c) {
  // 8 args: (neg_a, A, neg_b, B, c_mod, C, reuse_a, reuse_b)
  return __builtin_amdgcn_wmma_f32_16x16x32_bf16(false, a, false, b, (short)0, c,
                                                 false, false);
}
DEV bf16x16 cat16(bf16x8 lo, bf16x8 hi) {
  return __builtin_shufflevector(lo, hi, 0,1,2,3,4,5,6,7,8,9,10,11,12,13,14,15);
}
DEV f32x8 zero8() {
  f32x8 z;
#pragma unroll
  for (int i = 0; i < 8; i++) z[i] = 0.f;
  return z;
}
// branch-free tanh: clamp + single v_exp_f32 (avoids libm tanhf's EXEC dance)
DEV float fast_tanh(float x) {
  x = fminf(fmaxf(x, -15.f), 15.f);
  const float e = __expf(2.f * x);
  return (e - 1.f) / (e + 1.f);
}
DEV float gelu_f(float x) {  // tanh-approx GELU (jax.nn.gelu default)
  const float k0 = 0.7978845608028654f, k1 = 0.044715f;
  return 0.5f * x * (1.f + fast_tanh(k0 * (x + k1 * x * x * x)));
}

// ---------------------------------------------------------------------------
// fp32 -> bf16 conversion (weights)
// ---------------------------------------------------------------------------
__global__ void cvt_bf16_kernel(const float* __restrict__ in,
                                __bf16* __restrict__ out, int n) {
  int i = blockIdx.x * blockDim.x + threadIdx.x;
  if (i < n) out[i] = (__bf16)in[i];
}

// ---------------------------------------------------------------------------
// Block reduction helper (256 threads = 8 waves, wave32)
// ---------------------------------------------------------------------------
DEV float block_sum(float v, float* sred) {
#pragma unroll
  for (int o = 16; o > 0; o >>= 1) v += __shfl_xor(v, o);
  if ((threadIdx.x & 31) == 0) sred[threadIdx.x >> 5] = v;
  __syncthreads();
  float tot = 0.f;
#pragma unroll
  for (int i = 0; i < 8; i++) tot += sred[i];
  __syncthreads();
  return tot;
}

// ---------------------------------------------------------------------------
// Embedding gather + LayerNorm: h = LN(tok_emb[x[s]] + pos_emb[s])
// ---------------------------------------------------------------------------
__global__ __launch_bounds__(256)
void embed_ln_kernel(const int* __restrict__ x, const float* __restrict__ tok,
                     const float* __restrict__ pos, const float* __restrict__ g,
                     const float* __restrict__ bta, float* __restrict__ hf,
                     __bf16* __restrict__ hb) {
  __shared__ float sred[8];
  const int s = blockIdx.x, t = threadIdx.x;
  const size_t tb = (size_t)x[s] * 768, pb = (size_t)s * 768;
  float v0 = tok[tb + t]       + pos[pb + t];
  float v1 = tok[tb + t + 256] + pos[pb + t + 256];
  float v2 = tok[tb + t + 512] + pos[pb + t + 512];
  const float mean = block_sum(v0 + v1 + v2, sred) * (1.f / 768.f);
  v0 -= mean; v1 -= mean; v2 -= mean;
  const float var = block_sum(v0 * v0 + v1 * v1 + v2 * v2, sred) * (1.f / 768.f);
  const float rstd = rsqrtf(var + 1e-5f);
  const float o0 = v0 * rstd * g[t]       + bta[t];
  const float o1 = v1 * rstd * g[t + 256] + bta[t + 256];
  const float o2 = v2 * rstd * g[t + 512] + bta[t + 512];
  hf[pb + t] = o0; hf[pb + t + 256] = o1; hf[pb + t + 512] = o2;
  hb[pb + t] = (__bf16)o0; hb[pb + t + 256] = (__bf16)o1; hb[pb + t + 512] = (__bf16)o2;
}

// ---------------------------------------------------------------------------
// h' = LN(a + r): residual + LayerNorm (WRITE_B: also emit bf16 copy)
// ---------------------------------------------------------------------------
template <bool WRITE_B>
__global__ __launch_bounds__(256)
void ln_residual_kernel(const float* __restrict__ a, const float* __restrict__ r,
                        const float* __restrict__ g, const float* __restrict__ bta,
                        float* __restrict__ hf, __bf16* __restrict__ hb) {
  __shared__ float sred[8];
  const int s = blockIdx.x, t = threadIdx.x;
  const size_t pb = (size_t)s * 768;
  float v0 = a[pb + t]       + r[pb + t];
  float v1 = a[pb + t + 256] + r[pb + t + 256];
  float v2 = a[pb + t + 512] + r[pb + t + 512];
  const float mean = block_sum(v0 + v1 + v2, sred) * (1.f / 768.f);
  v0 -= mean; v1 -= mean; v2 -= mean;
  const float var = block_sum(v0 * v0 + v1 * v1 + v2 * v2, sred) * (1.f / 768.f);
  const float rstd = rsqrtf(var + 1e-5f);
  const float o0 = v0 * rstd * g[t]       + bta[t];
  const float o1 = v1 * rstd * g[t + 256] + bta[t + 256];
  const float o2 = v2 * rstd * g[t + 512] + bta[t + 512];
  hf[pb + t] = o0; hf[pb + t + 256] = o1; hf[pb + t + 512] = o2;
  if (WRITE_B) {
    hb[pb + t] = (__bf16)o0; hb[pb + t + 256] = (__bf16)o1; hb[pb + t + 512] = (__bf16)o2;
  }
}

// ---------------------------------------------------------------------------
// Tiled WMMA GEMM: C[M,N] = A[M,K](bf16) * B[K,N](bf16) + bias [+GELU].
// Block 256 thr = 8 waves (2x4); tile 128x128, BK=64, DOUBLE-BUFFERED LDS:
// one barrier per K-tile; next tile's async copies overlap current WMMAs.
// Each wave computes 64x32 (4x2 of 16x16). A staged row-major (async),
// B staged transposed so both fragment types are contiguous 16B LDS reads
// matching the ISA A/B VGPR layouts. Compile-time epilogue.
// ---------------------------------------------------------------------------
template <bool GELU, bool OUTF, bool OUTB>
__global__ __launch_bounds__(256)
void gemm_bf16_wmma(const __bf16* __restrict__ A, const __bf16* __restrict__ B,
                    const float* __restrict__ bias, float* __restrict__ Cf,
                    __bf16* __restrict__ Cb, int M, int N, int K) {
  __shared__ __align__(16) __bf16 sA[2][128][72];   // 2 x 128 rows x 64 k (+8 pad)
  __shared__ __align__(16) __bf16 sBT[2][128][72];  // 2 x 128 n    x 64 k (+8 pad)
  const int t = threadIdx.x;
  const int w = t >> 5, lane = t & 31, hl = lane >> 4, lm = lane & 15;
  const int wy = w >> 2, wx = w & 3;  // wave grid 2(M) x 4(N)
  const int bm = blockIdx.y * 128, bn = blockIdx.x * 128;
  const int ar = t >> 1, ac = (t & 1) * 32;  // A staging: 2 thr/row, 32 halves each
  const int br = t >> 2, bc = (t & 3) * 32;  // B staging: 4 thr/krow, 32 halves each
  (void)M;

  auto stageA = [&](int buf, int k0) {
    const __bf16* ga = A + (size_t)(bm + ar) * K + (k0 + ac);
#pragma unroll
    for (int i = 0; i < 4; i++) async_cp16(ga + i * 8, &sA[buf][ar][ac + i * 8]);
  };
  auto stageB = [&](int buf, int k0) {
    const __bf16* gb = B + (size_t)(k0 + br) * N + (bn + bc);
#pragma unroll
    for (int q = 0; q < 2; q++) {
      bf16x8 b0 = *(const bf16x8*)(gb + q * 16);
      bf16x8 b1 = *(const bf16x8*)(gb + q * 16 + 8);
#pragma unroll
      for (int i = 0; i < 8; i++) {
        sBT[buf][bc + q * 16 + i][br]     = b0[i];
        sBT[buf][bc + q * 16 + 8 + i][br] = b1[i];
      }
    }
  };

  f32x8 acc[4][2];
#pragma unroll
  for (int mt = 0; mt < 4; mt++)
#pragma unroll
    for (int nt = 0; nt < 2; nt++) acc[mt][nt] = zero8();

  stageA(0, 0);
  stageB(0, 0);
  int cur = 0;
  for (int k0 = 0; k0 < K; k0 += 64) {
    // Each wave waits its OWN async copies, then the barrier makes all waves'
    // copies visible and proves buf[cur^1] has no readers left.
    async_fence();
    __syncthreads();
    if (k0 + 64 < K) { stageA(cur ^ 1, k0 + 64); stageB(cur ^ 1, k0 + 64); }

#pragma unroll
    for (int ks = 0; ks < 2; ks++) {
      bf16x16 af[4], bfr[2];
#pragma unroll
      for (int mt = 0; mt < 4; mt++) {  // A frag: lane<16 K{0..7,16..23}, lane>=16 +8
        const __bf16* p = &sA[cur][wy * 64 + mt * 16 + lm][ks * 32 + hl * 8];
        af[mt] = cat16(*(const bf16x8*)p, *(const bf16x8*)(p + 16));
      }
#pragma unroll
      for (int nt = 0; nt < 2; nt++) {  // B frag: lane<16 K{0..7,16..23}, lane>=16 +8
        const __bf16* p = &sBT[cur][wx * 32 + nt * 16 + lm][ks * 32 + hl * 8];
        bfr[nt] = cat16(*(const bf16x8*)p, *(const bf16x8*)(p + 16));
      }
#pragma unroll
      for (int mt = 0; mt < 4; mt++)
#pragma unroll
        for (int nt = 0; nt < 2; nt++)
          acc[mt][nt] = wmma_bf16(af[mt], bfr[nt], acc[mt][nt]);
    }
    cur ^= 1;
  }

#pragma unroll
  for (int mt = 0; mt < 4; mt++)
#pragma unroll
    for (int nt = 0; nt < 2; nt++) {
      const int col = bn + wx * 32 + nt * 16 + lm;
      const float bv = bias[col];
#pragma unroll
      for (int j = 0; j < 8; j++) {  // C layout: row = (lane/16)*8 + j
        const int row = bm + wy * 64 + mt * 16 + hl * 8 + j;
        float v = acc[mt][nt][j] + bv;
        if (GELU) v = gelu_f(v);
        if (OUTF) Cf[(size_t)row * N + col] = v;
        if (OUTB) Cb[(size_t)row * N + col] = (__bf16)v;
      }
    }
}

// ---------------------------------------------------------------------------
// Sliding-window flash attention. Grid = H*C = 192 blocks, 256 thr = 8 waves.
// Wave handles 32 query rows; streams 12 key-blocks of 64 (3W = 768 keys).
// Scores = Q@K^T via WMMA, online softmax (row reductions within 16-lane
// halves, matching the C-matrix layout), O += P@V via WMMA after an LDS
// round-trip reshaping P from C-layout to A-layout. K-tile staged async.
// ---------------------------------------------------------------------------
__global__ __launch_bounds__(256)
void lf_attention_kernel(const __bf16* __restrict__ Q, const __bf16* __restrict__ Km,
                         const __bf16* __restrict__ Vm, const int* __restrict__ xids,
                         __bf16* __restrict__ Ob) {
  __shared__ __align__(16) __bf16 sK[64][72];     // key rows x DH (+pad)
  __shared__ __align__(16) __bf16 sVT[64][72];    // DH x key (+pad)  (V^T)
  __shared__ __align__(16) __bf16 sP[8][32][72];  // per-wave P staging

  const int hid = blockIdx.x >> 4;  // head
  const int c   = blockIdx.x & 15;  // chunk
  const int t = threadIdx.x, w = t >> 5, lane = t & 31, hl = lane >> 4, lm = lane & 15;
  const int qbase = w * 32;

  // Q fragments, register-resident for the whole kernel (2 m-tiles x 2 k-steps)
  bf16x16 qf[2][2];
#pragma unroll
  for (int mt = 0; mt < 2; mt++)
#pragma unroll
    for (int ks = 0; ks < 2; ks++) {
      const int s = c * 256 + qbase + mt * 16 + lm;
      const __bf16* p = Q + (size_t)s * 768 + hid * 64 + ks * 32 + hl * 8;
      qf[mt][ks] = cat16(*(const bf16x8*)p, *(const bf16x8*)(p + 16));
    }

  f32x8 oacc[2][4];
#pragma unroll
  for (int mt = 0; mt < 2; mt++)
#pragma unroll
    for (int od = 0; od < 4; od++) oacc[mt][od] = zero8();
  float mrun[2][8], lrun[2][8];
#pragma unroll
  for (int mt = 0; mt < 2; mt++)
#pragma unroll
    for (int j = 0; j < 8; j++) { mrun[mt][j] = -3.0e38f; lrun[mt][j] = 0.f; }

  const int kr = t >> 2, kd = (t & 3) * 16;  // K/V staging coords

  for (int kb = 0; kb < 12; kb++) {
    const int klb = kb * 64;
    const int gk = c * 256 + klb + kr - 256;  // global key position
    if (gk >= 0 && gk < 4096) {
      const __bf16* pk = Km + (size_t)gk * 768 + hid * 64 + kd;
      async_cp16(pk,     &sK[kr][kd]);
      async_cp16(pk + 8, &sK[kr][kd + 8]);
      const __bf16* pv = Vm + (size_t)gk * 768 + hid * 64 + kd;
      bf16x8 v0 = *(const bf16x8*)pv, v1 = *(const bf16x8*)(pv + 8);
#pragma unroll
      for (int i = 0; i < 8; i++) { sVT[kd + i][kr] = v0[i]; sVT[kd + 8 + i][kr] = v1[i]; }
    } else {
#pragma unroll
      for (int i = 0; i < 16; i++) { sK[kr][kd + i] = (__bf16)0.f; sVT[kd + i][kr] = (__bf16)0.f; }
    }
    async_fence();
    __syncthreads();

    // S = Q @ K^T  (B^T in LDS == K row-major)
    f32x8 sc[2][4];
#pragma unroll
    for (int mt = 0; mt < 2; mt++)
#pragma unroll
      for (int nt = 0; nt < 4; nt++) {
        f32x8 a = zero8();
#pragma unroll
        for (int ks = 0; ks < 2; ks++) {
          const __bf16* p = &sK[nt * 16 + lm][ks * 32 + hl * 8];
          a = wmma_bf16(qf[mt][ks], cat16(*(const bf16x8*)p, *(const bf16x8*)(p + 16)), a);
        }
        sc[mt][nt] = a;
      }

    // scale + mask: key padding, x!=0, |k_local - W - q_local| <= W
#pragma unroll
    for (int mt = 0; mt < 2; mt++)
#pragma unroll
      for (int nt = 0; nt < 4; nt++) {
        const int kl = klb + nt * 16 + lm;
        const int gk2 = c * 256 + kl - 256;
        const bool kvalid = (gk2 >= 0) && (gk2 < 4096) && (xids[gk2 & 4095] != 0);
#pragma unroll
        for (int j = 0; j < 8; j++) {
          const int ql = qbase + mt * 16 + hl * 8 + j;
          const int rel = kl - 256 - ql;
          const bool ok = kvalid && (rel >= -256) && (rel <= 256);
          sc[mt][nt][j] = ok ? sc[mt][nt][j] * 0.125f : -1.0e9f;
        }
      }

    // online softmax: rows live across the 16 lanes of a half-wave
#pragma unroll
    for (int mt = 0; mt < 2; mt++)
#pragma unroll
      for (int j = 0; j < 8; j++) {
        float bmx = sc[mt][0][j];
#pragma unroll
        for (int nt = 1; nt < 4; nt++) bmx = fmaxf(bmx, sc[mt][nt][j]);
#pragma unroll
        for (int o = 8; o > 0; o >>= 1) bmx = fmaxf(bmx, __shfl_xor(bmx, o));
        const float mn = fmaxf(mrun[mt][j], bmx);
        const float sf = __expf(mrun[mt][j] - mn);
        mrun[mt][j] = mn;
        float bs = 0.f;
#pragma unroll
        for (int nt = 0; nt < 4; nt++) {
          const float pv = __expf(sc[mt][nt][j] - mn);
          sc[mt][nt][j] = pv; bs += pv;
        }
#pragma unroll
        for (int o = 8; o > 0; o >>= 1) bs += __shfl_xor(bs, o);
        lrun[mt][j] = lrun[mt][j] * sf + bs;
#pragma unroll
        for (int od = 0; od < 4; od++) oacc[mt][od][j] *= sf;
      }

    // stage P (C-layout -> LDS row-major)
#pragma unroll
    for (int mt = 0; mt < 2; mt++)
#pragma unroll
      for (int nt = 0; nt < 4; nt++)
#pragma unroll
        for (int j = 0; j < 8; j++)
          sP[w][mt * 16 + hl * 8 + j][nt * 16 + lm] = (__bf16)sc[mt][nt][j];
    __syncthreads();

    // O += P @ V  (A frags from sP, B^T in LDS == V^T)
#pragma unroll
    for (int mt = 0; mt < 2; mt++)
#pragma unroll
      for (int ks = 0; ks < 2; ks++) {
        const __bf16* pa = &sP[w][mt * 16 + lm][ks * 32 + hl * 8];
        bf16x16 af = cat16(*(const bf16x8*)pa, *(const bf16x8*)(pa + 16));
#pragma unroll
        for (int od = 0; od < 4; od++) {
          const __bf16* pb = &sVT[od * 16 + lm][ks * 32 + hl * 8];
          oacc[mt][od] =
              wmma_bf16(af, cat16(*(const bf16x8*)pb, *(const bf16x8*)(pb + 16)), oacc[mt][od]);
        }
      }
    __syncthreads();  // protect sK/sVT/sP before next block's staging
  }

  // normalize + write (B,S,D) with head-interleaved columns
#pragma unroll
  for (int mt = 0; mt < 2; mt++)
#pragma unroll
    for (int j = 0; j < 8; j++) {
      const float inv = lrun[mt][j] > 0.f ? 1.f / lrun[mt][j] : 0.f;
      const int s = c * 256 + qbase + mt * 16 + hl * 8 + j;
#pragma unroll
      for (int od = 0; od < 4; od++) {
        const int col = hid * 64 + od * 16 + lm;
        Ob[(size_t)s * 768 + col] = (__bf16)(oacc[mt][od][j] * inv);
      }
    }
}

// ---------------------------------------------------------------------------
// Pooler: out = tanh(h2[0,:] @ Wp + bp) -- only row 0 of h2 is ever needed.
// ---------------------------------------------------------------------------
__global__ __launch_bounds__(256)
void pooler_kernel(const float* __restrict__ h2, const float* __restrict__ Wp,
                   const float* __restrict__ bp, float* __restrict__ out) {
  const int n = blockIdx.x * blockDim.x + threadIdx.x;
  if (n >= 768) return;
  float a = bp[n];
  for (int k = 0; k < 768; k++) a += h2[k] * Wp[(size_t)k * 768 + n];
  out[n] = fast_tanh(a);
}

// ---------------------------------------------------------------------------
// Host-side orchestration
// ---------------------------------------------------------------------------
static inline void* wsa(char*& p, size_t bytes) {
  void* r = (void*)p;
  p += (bytes + 255) & ~(size_t)255;
  return r;
}

extern "C" void kernel_launch(void* const* d_in, const int* in_sizes, int n_in,
                              void* d_out, int out_size, void* d_ws, size_t ws_size,
                              hipStream_t stream) {
  (void)in_sizes; (void)n_in; (void)out_size; (void)ws_size;
  const int*   x     = (const int*)d_in[0];
  const float* tok   = (const float*)d_in[1];
  const float* pos   = (const float*)d_in[2];
  const float* ln_eg = (const float*)d_in[3];
  const float* ln_eb = (const float*)d_in[4];
  const float* Wq = (const float*)d_in[5];  const float* bq = (const float*)d_in[6];
  const float* Wk = (const float*)d_in[7];  const float* bk = (const float*)d_in[8];
  const float* Wv = (const float*)d_in[9];  const float* bv = (const float*)d_in[10];
  const float* Wo = (const float*)d_in[11]; const float* bo = (const float*)d_in[12];
  const float* ln1g = (const float*)d_in[13]; const float* ln1b = (const float*)d_in[14];
  const float* W1 = (const float*)d_in[15]; const float* b1 = (const float*)d_in[16];
  const float* W2 = (const float*)d_in[17]; const float* b2 = (const float*)d_in[18];
  const float* ln2g = (const float*)d_in[19]; const float* ln2b = (const float*)d_in[20];
  const float* Wp = (const float*)d_in[21]; const float* bp = (const float*)d_in[22];

  const size_t S = 4096, D = 768, DFF = 3072;
  char* p = (char*)d_ws;
  __bf16* wq_b = (__bf16*)wsa(p, D * D * 2);
  __bf16* wk_b = (__bf16*)wsa(p, D * D * 2);
  __bf16* wv_b = (__bf16*)wsa(p, D * D * 2);
  __bf16* wo_b = (__bf16*)wsa(p, D * D * 2);
  __bf16* w1_b = (__bf16*)wsa(p, D * DFF * 2);
  __bf16* w2_b = (__bf16*)wsa(p, DFF * D * 2);
  float*  h_f  = (float*)wsa(p, S * D * 4);
  __bf16* h_b  = (__bf16*)wsa(p, S * D * 2);
  __bf16* q_b  = (__bf16*)wsa(p, S * D * 2);
  __bf16* k_b  = (__bf16*)wsa(p, S * D * 2);
  __bf16* v_b  = (__bf16*)wsa(p, S * D * 2);
  __bf16* at_b = (__bf16*)wsa(p, S * D * 2);
  float*  t0_f = (float*)wsa(p, S * D * 4);
  float*  h1_f = (float*)wsa(p, S * D * 4);
  __bf16* h1_b = (__bf16*)wsa(p, S * D * 2);
  __bf16* g_b  = (__bf16*)wsa(p, S * DFF * 2);
  float*  f_f  = (float*)wsa(p, S * D * 4);
  float*  h2_f = (float*)wsa(p, S * D * 4);

  // 1) weights -> bf16
  const int ndd = (int)(D * D), ndf = (int)(D * DFF);
  cvt_bf16_kernel<<<(ndd + 255) / 256, 256, 0, stream>>>(Wq, wq_b, ndd);
  cvt_bf16_kernel<<<(ndd + 255) / 256, 256, 0, stream>>>(Wk, wk_b, ndd);
  cvt_bf16_kernel<<<(ndd + 255) / 256, 256, 0, stream>>>(Wv, wv_b, ndd);
  cvt_bf16_kernel<<<(ndd + 255) / 256, 256, 0, stream>>>(Wo, wo_b, ndd);
  cvt_bf16_kernel<<<(ndf + 255) / 256, 256, 0, stream>>>(W1, w1_b, ndf);
  cvt_bf16_kernel<<<(ndf + 255) / 256, 256, 0, stream>>>(W2, w2_b, ndf);

  // 2) embedding + LN
  embed_ln_kernel<<<4096, 256, 0, stream>>>(x, tok, pos, ln_eg, ln_eb, h_f, h_b);

  // 3) Q/K/V projections (bf16 out only)
  dim3 gqkv(6, 32);  // N/128 x M/128
  gemm_bf16_wmma<false, false, true><<<gqkv, 256, 0, stream>>>(h_b, wq_b, bq, nullptr, q_b, 4096, 768, 768);
  gemm_bf16_wmma<false, false, true><<<gqkv, 256, 0, stream>>>(h_b, wk_b, bk, nullptr, k_b, 4096, 768, 768);
  gemm_bf16_wmma<false, false, true><<<gqkv, 256, 0, stream>>>(h_b, wv_b, bv, nullptr, v_b, 4096, 768, 768);

  // 4) sliding-window attention (H*C = 192 workgroups)
  lf_attention_kernel<<<192, 256, 0, stream>>>(q_b, k_b, v_b, x, at_b);

  // 5) output projection + residual LN
  gemm_bf16_wmma<false, true, false><<<gqkv, 256, 0, stream>>>(at_b, wo_b, bo, t0_f, nullptr, 4096, 768, 768);
  ln_residual_kernel<true><<<4096, 256, 0, stream>>>(h_f, t0_f, ln1g, ln1b, h1_f, h1_b);

  // 6) FFN: GELU(h1 @ W1 + b1) @ W2 + b2, residual LN
  dim3 gff1(24, 32);
  gemm_bf16_wmma<true, false, true><<<gff1, 256, 0, stream>>>(h1_b, w1_b, b1, nullptr, g_b, 4096, 3072, 768);
  gemm_bf16_wmma<false, true, false><<<gqkv, 256, 0, stream>>>(g_b, w2_b, b2, f_f, nullptr, 4096, 768, 3072);
  ln_residual_kernel<false><<<4096, 256, 0, stream>>>(h1_f, f_f, ln2g, ln2b, h2_f, nullptr);

  // 7) pooler: tanh(h2[0] @ Wp + bp) -> d_out (768 f32)
  pooler_kernel<<<3, 256, 0, stream>>>(h2_f, Wp, bp, (float*)d_out);
}
